// LCMCOV_49143015801068
// MI455X (gfx1250) — matrix operand order repaired and checked by reading, not compiled, not run on hardware
//
#include <hip/hip_runtime.h>
#include <hip/hip_bf16.h>

// Problem constants (from reference)
#define B_    64
#define C_    2048
#define M_    196      // H*W
#define MP_   224      // M padded to multiple of 32 (7 pairs of 16-tiles)
#define DR_   256
#define OUTN_ 32896    // DR*(DR+1)/2

typedef __bf16 bf16_t;
typedef __attribute__((ext_vector_type(16))) bf16_t       bf16x16;
typedef __attribute__((ext_vector_type(8)))  float        v8f;
typedef __attribute__((ext_vector_type(2)))  float        v2f;
typedef __attribute__((ext_vector_type(4)))  unsigned int u32x4;

union ABFrag { bf16x16 v; u32x4 q[2]; };

// ---------------------------------------------------------------------------
// 1) Convert conv weight f32 -> bf16 (row-major [DR][C])
// ---------------------------------------------------------------------------
__global__ void k_convert_w(const float* __restrict__ w, bf16_t* __restrict__ wb, int n) {
    int i = blockIdx.x * blockDim.x + threadIdx.x;
    if (i < n) wb[i] = (bf16_t)w[i];
}

// ---------------------------------------------------------------------------
// 2) Transpose x [B][C][M] f32 -> Xt [B][MP][C] bf16, zero-fill rows 196..223
//    (gives K-contiguous per-column data so WMMA B fragments are b128 loads)
// ---------------------------------------------------------------------------
__global__ void k_transpose_x(const float* __restrict__ x, bf16_t* __restrict__ xt) {
    __shared__ float tile[32][33];
    int b  = blockIdx.z;
    int m0 = blockIdx.x * 32;
    int c0 = blockIdx.y * 32;
    int tx = threadIdx.x, ty = threadIdx.y;
#pragma unroll
    for (int i = 0; i < 4; ++i) {
        int c = c0 + ty + i * 8;
        int m = m0 + tx;
        float v = 0.0f;
        if (m < M_) v = x[((size_t)b * C_ + c) * M_ + m];
        tile[ty + i * 8][tx] = v;
    }
    __syncthreads();
#pragma unroll
    for (int i = 0; i < 4; ++i) {
        int m = m0 + ty + i * 8;
        int c = c0 + tx;
        if (m < MP_) xt[((size_t)b * MP_ + m) * C_ + c] = (bf16_t)tile[tx][ty + i * 8];
    }
}

// ---------------------------------------------------------------------------
// 3) Conv GEMM: Y[b] = W(256x2048) * X[b](2048x196), bf16 WMMA, f32 acc,
//    fused BN(eval)+ReLU epilogue. Block = 8 waves. Wave wv owns a 32x32
//    register tile: DR rows [32wv, 32wv+32) x M cols [32*bx, 32*bx+32)
//    => 2x2 WMMA tiles, 4 accumulators, 4 WMMAs per K-step of 32.
// ---------------------------------------------------------------------------
__global__ void __launch_bounds__(256) k_conv_gemm(
    const bf16_t* __restrict__ wb, const bf16_t* __restrict__ xt,
    const float* __restrict__ gamma, const float* __restrict__ beta,
    const float* __restrict__ rmean, const float* __restrict__ rvar,
    float* __restrict__ yc) {
    int b    = blockIdx.z;
    int m0   = blockIdx.x * 32;
    int wv   = threadIdx.x >> 5;
    int lane = threadIdx.x & 31;
    int half = lane >> 4;
    int n    = lane & 15;
    int d0   = wv * 32;

    const bf16_t* wr0 = wb + (size_t)(d0 + n) * C_;
    const bf16_t* wr1 = wb + (size_t)(d0 + 16 + n) * C_;
    const bf16_t* xr0 = xt + ((size_t)b * MP_ + m0 + n) * C_;
    const bf16_t* xr1 = xt + ((size_t)b * MP_ + m0 + 16 + n) * C_;

    v8f acc00 = {0.f, 0.f, 0.f, 0.f, 0.f, 0.f, 0.f, 0.f};
    v8f acc01 = {0.f, 0.f, 0.f, 0.f, 0.f, 0.f, 0.f, 0.f};
    v8f acc10 = {0.f, 0.f, 0.f, 0.f, 0.f, 0.f, 0.f, 0.f};
    v8f acc11 = {0.f, 0.f, 0.f, 0.f, 0.f, 0.f, 0.f, 0.f};
    const int alo = half * 16;  // A fragment lo-run K offset within 32-wide tile
    const int blo = half * 16;  // B fragment K offset within 32-wide tile

    for (int c0 = 0; c0 < C_; c0 += 32) {
        __builtin_prefetch(xr0 + c0 + 256, 0, 1);  // global_prefetch_b8
        __builtin_prefetch(xr1 + c0 + 256, 0, 1);
        ABFrag a0, a1, b0, b1;
        a0.q[0] = *(const u32x4*)(const void*)(wr0 + c0 + alo);
        a0.q[1] = *(const u32x4*)(const void*)(wr0 + c0 + alo + 8);
        a1.q[0] = *(const u32x4*)(const void*)(wr1 + c0 + alo);
        a1.q[1] = *(const u32x4*)(const void*)(wr1 + c0 + alo + 8);
        b0.q[0] = *(const u32x4*)(const void*)(xr0 + c0 + blo);
        b0.q[1] = *(const u32x4*)(const void*)(xr0 + c0 + blo + 8);
        b1.q[0] = *(const u32x4*)(const void*)(xr1 + c0 + blo);
        b1.q[1] = *(const u32x4*)(const void*)(xr1 + c0 + blo + 8);
        acc00 = __builtin_amdgcn_wmma_f32_16x16x32_bf16(false, a0.v, false, b0.v,
                                                        (short)0, acc00, false, false);
        acc01 = __builtin_amdgcn_wmma_f32_16x16x32_bf16(false, a0.v, false, b1.v,
                                                        (short)0, acc01, false, false);
        acc10 = __builtin_amdgcn_wmma_f32_16x16x32_bf16(false, a1.v, false, b0.v,
                                                        (short)0, acc10, false, false);
        acc11 = __builtin_amdgcn_wmma_f32_16x16x32_bf16(false, a1.v, false, b1.v,
                                                        (short)0, acc11, false, false);
    }

    int col0 = m0 + n;
    int col1 = m0 + 16 + n;
#pragma unroll
    for (int i = 0; i < 8; ++i) {
        int r0 = d0 + i + half * 8;
        float s0  = gamma[r0] / sqrtf(rvar[r0] + 1e-5f);
        float bb0 = beta[r0] - rmean[r0] * s0;
        float v;
        v = fmaf(acc00[i], s0, bb0);
        yc[((size_t)b * DR_ + r0) * MP_ + col0] = v > 0.f ? v : 0.f;
        v = fmaf(acc01[i], s0, bb0);
        yc[((size_t)b * DR_ + r0) * MP_ + col1] = v > 0.f ? v : 0.f;

        int r1 = r0 + 16;
        float s1  = gamma[r1] / sqrtf(rvar[r1] + 1e-5f);
        float bb1 = beta[r1] - rmean[r1] * s1;
        v = fmaf(acc10[i], s1, bb1);
        yc[((size_t)b * DR_ + r1) * MP_ + col0] = v > 0.f ? v : 0.f;
        v = fmaf(acc11[i], s1, bb1);
        yc[((size_t)b * DR_ + r1) * MP_ + col1] = v > 0.f ? v : 0.f;
    }
}

// ---------------------------------------------------------------------------
// 4) Row-center (fp64 mean over M=196), zero the MP pad columns
// ---------------------------------------------------------------------------
__global__ void __launch_bounds__(256) k_center(float* __restrict__ yc) {
    int d = blockIdx.x, b = blockIdx.y, t = threadIdx.x;
    float* row = yc + ((size_t)b * DR_ + d) * MP_;
    __shared__ double red[256];
    float v = (t < M_) ? row[t] : 0.f;
    red[t] = (double)v;
    __syncthreads();
    for (int s = 128; s > 0; s >>= 1) {
        if (t < s) red[t] += red[t + s];
        __syncthreads();
    }
    double mean = red[0] / (double)M_;
    if (t < MP_) row[t] = (t < M_) ? (float)((double)v - mean) : 0.f;
}

// ---------------------------------------------------------------------------
// 5) Gram: cov[b] = xc * xc^T / M + eps*I using fp32 WMMA (16x16x4 f32).
//    Wave -> one 16x16 cov tile; A rows from xc[i0+..], B cols from xc[j0+..].
// ---------------------------------------------------------------------------
__global__ void __launch_bounds__(256) k_gram(const float* __restrict__ xc,
                                              double* __restrict__ cov) {
    int b    = blockIdx.z;
    int i0   = blockIdx.x * 16;
    int wv   = threadIdx.x >> 5;
    int lane = threadIdx.x & 31;
    int half = lane >> 4;
    int n    = lane & 15;
    int j0   = (blockIdx.y * 8 + wv) * 16;

    const float* ar = xc + ((size_t)b * DR_ + i0 + n) * MP_;
    const float* br = xc + ((size_t)b * DR_ + j0 + n) * MP_;

    v8f acc = {0.f, 0.f, 0.f, 0.f, 0.f, 0.f, 0.f, 0.f};
    for (int m = 0; m < MP_; m += 4) {
        v2f av = *(const v2f*)(const void*)(ar + m + half * 2);
        v2f bv = *(const v2f*)(const void*)(br + m + half * 2);
        acc = __builtin_amdgcn_wmma_f32_16x16x4_f32(false, av, false, bv,
                                                    (short)0, acc, false, false);
    }
    const double inv_m = 1.0 / (double)M_;
#pragma unroll
    for (int i = 0; i < 8; ++i) {
        int ii = i0 + i + half * 8;
        int jj = j0 + n;
        double v = (double)acc[i] * inv_m + (ii == jj ? 1e-6 : 0.0);
        cov[((size_t)b * DR_ + ii) * DR_ + jj] = v;
    }
}

// ---------------------------------------------------------------------------
// 6) fp64 Cholesky, one block (256 thr) per batch, right-looking, in place.
//    Thread t owns row t; pivot column cached in LDS.
// ---------------------------------------------------------------------------
__global__ void __launch_bounds__(256) k_cholesky(double* __restrict__ cov) {
    int b = blockIdx.x, t = threadIdx.x;
    double* A = cov + (size_t)b * DR_ * DR_;
    __shared__ double colk[DR_];
    __shared__ double sdk;
    for (int k = 0; k < DR_; ++k) {
        __syncthreads();
        if (t == 0) sdk = sqrt(A[(size_t)k * DR_ + k]);
        __syncthreads();
        double lik = 0.0;
        if (t > k)       lik = A[(size_t)t * DR_ + k] / sdk;
        else if (t == k) lik = sdk;
        colk[t] = lik;
        if (t >= k) A[(size_t)t * DR_ + k] = lik;
        __syncthreads();
        for (int j = k + 1; j <= t; ++j)
            A[(size_t)t * DR_ + j] -= lik * colk[j];
    }
}

// ---------------------------------------------------------------------------
// 7) Output: upper-tri row-major of sym = tril(L,-1)+diag(log L_ii) + its T
//    => out(i,j) = (i==j) ? 2*log(L_ii) : L[j][i]   (cast to f32)
// ---------------------------------------------------------------------------
__global__ void __launch_bounds__(256) k_output(const double* __restrict__ cov,
                                                float* __restrict__ out) {
    int i = blockIdx.x, b = blockIdx.y, j = threadIdx.x;
    if (j < i) return;
    const double* A = cov + (size_t)b * DR_ * DR_;
    size_t base = (size_t)b * OUTN_ + (size_t)i * DR_ - (size_t)(i * (i - 1) / 2);
    double v = (i == j) ? 2.0 * log(A[(size_t)i * DR_ + i])
                        : A[(size_t)j * DR_ + i];
    out[base + (size_t)(j - i)] = (float)v;
}

// ---------------------------------------------------------------------------
extern "C" void kernel_launch(void* const* d_in, const int* in_sizes, int n_in,
                              void* d_out, int out_size, void* d_ws, size_t ws_size,
                              hipStream_t stream) {
    const float* x     = (const float*)d_in[0];
    const float* wconv = (const float*)d_in[1];
    const float* gamma = (const float*)d_in[2];
    const float* beta  = (const float*)d_in[3];
    const float* rmean = (const float*)d_in[4];
    const float* rvar  = (const float*)d_in[5];
    float* out = (float*)d_out;
    (void)in_sizes; (void)n_in; (void)out_size; (void)ws_size;

    char* ws = (char*)d_ws;
    size_t off = 0;
    auto take = [&](size_t bytes) {
        char* p = ws + off;
        off = (off + bytes + 255) & ~(size_t)255;
        return p;
    };
    bf16_t* xt  = (bf16_t*)take((size_t)B_ * MP_ * C_ * sizeof(bf16_t)); // 58.7 MB
    bf16_t* wb  = (bf16_t*)take((size_t)DR_ * C_ * sizeof(bf16_t));      //  1.0 MB
    float*  yc  = (float*)take((size_t)B_ * DR_ * MP_ * sizeof(float));  // 14.7 MB
    double* cov = (double*)take((size_t)B_ * DR_ * DR_ * sizeof(double));// 33.5 MB

    k_convert_w<<<dim3((DR_ * C_) / 256), dim3(256), 0, stream>>>(wconv, wb, DR_ * C_);
    k_transpose_x<<<dim3(MP_ / 32, C_ / 32, B_), dim3(32, 8), 0, stream>>>(x, xt);
    k_conv_gemm<<<dim3(MP_ / 32, 1, B_), dim3(256), 0, stream>>>(wb, xt, gamma, beta,
                                                                 rmean, rvar, yc);
    k_center<<<dim3(DR_, B_), dim3(256), 0, stream>>>(yc);
    k_gram<<<dim3(DR_ / 16, 2, B_), dim3(256), 0, stream>>>(yc, cov);
    k_cholesky<<<dim3(B_), dim3(256), 0, stream>>>(cov);
    k_output<<<dim3(DR_, B_), dim3(256), 0, stream>>>(cov, out);
}
